// GraphEncoder_48481590837644
// MI455X (gfx1250) — matrix operand order repaired and checked by reading, not compiled, not run on hardware
//
#include <hip/hip_runtime.h>
#include <hip/hip_bf16.h>
#include <stdint.h>

// ---------------------------------------------------------------------------
// 2-layer GAT for MI455X (gfx1250, wave32, WMMA).
//   N=20000 nodes, E=320000 edges (+N self loops), IN_F=128, HID=64, HEADS=4
// GEMMs: fp32 -> packed-bf16 pre-pass, then pure-WMMA GEMM
// (v_wmma_f32_16x16x32_bf16, 5 WMMAs/wave/K-step, fp32 accumulate).
// Edge softmax + scatter via float atomics (memory-bound, ~0.7GB -> ~30us).
// ---------------------------------------------------------------------------

#define GAT_N      20000
#define GAT_E      320000
#define GAT_NE     (GAT_E + GAT_N)   // with self loops
#define GAT_INF    128
#define GAT_HID    64
#define GAT_HEADS  4
#define NEG_SLOPE  0.2f

typedef __attribute__((ext_vector_type(16))) __bf16 v16bf;
typedef __attribute__((ext_vector_type(8)))  float  v8f;
typedef __attribute__((ext_vector_type(8)))  int    v8i;

// ---- bf16 pack helpers -----------------------------------------------------
__device__ __forceinline__ unsigned pack_bf16(float a, float b) {
  const unsigned short ua = __builtin_bit_cast(unsigned short, (__bf16)a);
  const unsigned short ub = __builtin_bit_cast(unsigned short, (__bf16)b);
  return (unsigned)ua | ((unsigned)ub << 16);
}

// ---- monotone float <-> int encoding for atomicMax on signed ints ----------
__device__ __forceinline__ int enc_f32(float x) {
  int b = __float_as_int(x);
  return b >= 0 ? b : (b ^ 0x7FFFFFFF);
}
__device__ __forceinline__ float dec_f32(int k) {
  int b = k >= 0 ? k : (k ^ 0x7FFFFFFF);
  return __int_as_float(b);
}
#define ENC_NEG_INF ((int)0x807FFFFF)   // enc_f32(-inf)

__device__ __forceinline__ void edge_nodes(const int* __restrict__ ei, int e,
                                           int& s, int& d) {
  if (e < GAT_E) { s = ei[e]; d = ei[GAT_E + e]; }
  else           { s = d = e - GAT_E; }
}

// ---------------------------------------------------------------------------
// Pre-pass 1: fp32 row-major -> bf16 packed K-pairs (dwords), same layout.
// Apk[m][kk] = pack(A[m][2kk], A[m][2kk+1])     (npairs = M*K/2)
// ---------------------------------------------------------------------------
__global__ void convert_bf16_pairs(const float* __restrict__ in,
                                   unsigned* __restrict__ out,
                                   long long npairs) {
  const long long t = (long long)blockIdx.x * blockDim.x + threadIdx.x;
  if (t >= npairs) return;
  const float2 v = *(const float2*)(in + 2 * t);
  out[t] = pack_bf16(v.x, v.y);
}

// ---------------------------------------------------------------------------
// Pre-pass 2: weight fp32 [K,N] -> B-fragment-ready swizzle:
// Wp[kk*N + n] = pack(W[2kk][n], W[2kk+1][n])   (kk = 0..K/2-1)
// GEMM B-tiles then stage as straight dword copies.
// ---------------------------------------------------------------------------
__global__ void convert_pack_w(const float* __restrict__ W,
                               unsigned* __restrict__ Wp, int K, int N) {
  const int t = blockIdx.x * blockDim.x + threadIdx.x;
  if (t >= (K >> 1) * N) return;
  const int kk = t / N;
  const int n  = t - kk * N;
  Wp[t] = pack_bf16(W[(2 * kk) * N + n], W[(2 * kk + 1) * N + n]);
}

// ---------------------------------------------------------------------------
// WMMA GEMM: C[M,N] = A @ B, packed-bf16 inputs, fp32 accumulate.
// Block = 128 threads (4 wave32). Block tile 80(M) x 64(N), K-step 32.
// Wave w owns 16-col strip w; runs 5 WMMAs per K-step reusing one B fragment.
// M % 80 == 0, N % 64 == 0, K % 32 == 0 for this problem (20000, 256/64, ...).
// ---------------------------------------------------------------------------
__global__ __launch_bounds__(128)
void gemm_wmma_bf16pk(const unsigned* __restrict__ Apk,  // [M][K/2] dwords
                      const unsigned* __restrict__ Bpk,  // [K/2][N] dwords
                      float* __restrict__ C, int M, int N, int K) {
  __shared__ unsigned As[80][18];   // stride 18: conflict-free fragment gather
  __shared__ unsigned Bs[16][66];   // stride 66: half-waves hit disjoint banks

  const int tid  = threadIdx.x;
  const int wave = tid >> 5;        // 0..3  -> 16-col strip
  const int lane = tid & 31;
  const int half = lane >> 4;       // wave32 half (ISA 7.12.2 layouts)
  const int mr   = lane & 15;
  const int m0   = blockIdx.x * 80;
  const int n0   = blockIdx.y * 64;
  const int Kd   = K >> 1;          // dwords per A row

  v8f acc[5] = {};

  for (int k0 = 0; k0 < Kd; k0 += 16) {   // 16 dwords = 32 bf16 K-slice
    // -- stage A tile: 80 rows x 16 dwords (coalesced dword stream) --
#pragma unroll
    for (int i = 0; i < 10; ++i) {
      const int idx = tid + 128 * i;        // 0..1279
      const int m   = idx >> 4;
      const int kk  = idx & 15;
      As[m][kk] = Apk[(size_t)(m0 + m) * Kd + k0 + kk];
    }
    // -- stage B tile: 16 rows x 64 dwords (already fragment-swizzled) --
#pragma unroll
    for (int i = 0; i < 8; ++i) {
      const int idx = tid + 128 * i;        // 0..1023
      const int kk  = idx >> 6;
      const int n   = idx & 63;
      Bs[kk][n] = Bpk[(size_t)(k0 + kk) * N + n0 + n];
    }
    __syncthreads();

    // -- B fragment (contiguous k, same as A: lanes 0-15 K=0..15, lanes 16-31 K=16..31) --
    v8i bi;
#pragma unroll
    for (int v = 0; v < 8; ++v) bi[v] = (int)Bs[8 * half + v][(wave << 4) + mr];
    const v16bf fb = __builtin_bit_cast(v16bf, bi);

    // -- 5 M-tiles, one WMMA each, fb reused --
#pragma unroll
    for (int mt = 0; mt < 5; ++mt) {
      v8i ai;
#pragma unroll
      for (int v = 0; v < 8; ++v) {
        // A dword v holds K-pair kk = v + 8*half
        const int kk = v + 8 * half;
        ai[v] = (int)As[mt * 16 + mr][kk];
      }
      const v16bf fa = __builtin_bit_cast(v16bf, ai);
      acc[mt] = __builtin_amdgcn_wmma_f32_16x16x32_bf16(
          false, fa, false, fb, (short)0, acc[mt], false, false);
    }
    __syncthreads();
  }

  // -- store: C VGPR r -> row r (lanes 0-15) / r+8 (lanes 16-31) --
  const int ng = n0 + (wave << 4) + mr;
#pragma unroll
  for (int mt = 0; mt < 5; ++mt)
#pragma unroll
    for (int r = 0; r < 8; ++r)
      C[(size_t)(m0 + mt * 16 + r + 8 * half) * N + ng] = acc[mt][r];
}

// ---------------------------------------------------------------------------
// Per-node half scores: al_src[n,h] = dot(h[n,h,:], a_src[h,:]) (and dst).
// One wave32 per (node, head); shuffle-tree reduction.
// ---------------------------------------------------------------------------
__global__ void half_scores(const float* __restrict__ h,
                            const float* __restrict__ a_src,
                            const float* __restrict__ a_dst,
                            float* __restrict__ al_src,
                            float* __restrict__ al_dst,
                            int HC, int C) {
  const int n    = blockIdx.x;
  const int head = threadIdx.x >> 5;
  const int lane = threadIdx.x & 31;
  const float* row = h + (size_t)n * HC + head * C;
  const float* as  = a_src + head * C;
  const float* ad  = a_dst + head * C;
  float ps = 0.f, pd = 0.f;
  for (int c = lane; c < C; c += 32) {
    const float v = row[c];
    ps += v * as[c];
    pd += v * ad[c];
  }
#pragma unroll
  for (int off = 16; off > 0; off >>= 1) {
    ps += __shfl_xor(ps, off, 32);
    pd += __shfl_xor(pd, off, 32);
  }
  if (lane == 0) {
    const int H = HC / C;
    al_src[n * H + head] = ps;
    al_dst[n * H + head] = pd;
  }
}

// ---------------------------------------------------------------------------
__global__ void fill_int(int* __restrict__ p, int n, int val) {
  const int t = blockIdx.x * blockDim.x + threadIdx.x;
  if (t < n) p[t] = val;
}

// segment max of leaky-relu scores over destination nodes
__global__ void edge_max_kernel(const int* __restrict__ ei,
                                const float* __restrict__ al_src,
                                const float* __restrict__ al_dst,
                                int* __restrict__ emax, int H) {
  const int t = blockIdx.x * blockDim.x + threadIdx.x;
  if (t >= GAT_NE * H) return;
  const int e  = t / H;
  const int hh = t - e * H;
  int s, d; edge_nodes(ei, e, s, d);
  float sc = al_src[s * H + hh] + al_dst[d * H + hh];
  sc = sc > 0.f ? sc : NEG_SLOPE * sc;
  atomicMax(&emax[d * H + hh], enc_f32(sc));
}

// exp(score - max) + segment sum; prefetch h[src] rows into L2 for aggregate
__global__ void edge_exp_kernel(const int* __restrict__ ei,
                                const float* __restrict__ al_src,
                                const float* __restrict__ al_dst,
                                const int* __restrict__ emax,
                                float* __restrict__ eexp,
                                float* __restrict__ den,
                                const float* __restrict__ hfeat,
                                int H, int HC) {
  const int t = blockIdx.x * blockDim.x + threadIdx.x;
  if (t >= GAT_NE * H) return;
  const int e  = t / H;
  const int hh = t - e * H;
  int s, d; edge_nodes(ei, e, s, d);
  float sc = al_src[s * H + hh] + al_dst[d * H + hh];
  sc = sc > 0.f ? sc : NEG_SLOPE * sc;
  const float ex = __expf(sc - dec_f32(emax[d * H + hh]));
  eexp[(size_t)e * H + hh] = ex;
  atomicAdd(&den[d * H + hh], ex);
  // warm L2 with the gather row the aggregate pass will read
  __builtin_prefetch(hfeat + (size_t)s * HC + hh * (HC / H), 0, 0);
}

// out[dst, h, c] += h[src, h, c] * alpha[e, h]
__global__ void aggregate_kernel(const int* __restrict__ ei,
                                 const float* __restrict__ hfeat,
                                 const float* __restrict__ eexp,
                                 const float* __restrict__ den,
                                 float* __restrict__ agg, int H, int C) {
  const long long t = (long long)blockIdx.x * blockDim.x + threadIdx.x;
  const int HC = H * C;
  const long long total = (long long)GAT_NE * HC;
  if (t >= total) return;
  const int e   = (int)(t / HC);
  const int rem = (int)(t - (long long)e * HC);
  const int hh  = rem / C;
  int s, d; edge_nodes(ei, e, s, d);
  const float alpha = eexp[(size_t)e * H + hh] / (den[d * H + hh] + 1e-16f);
  atomicAdd(&agg[(size_t)d * HC + rem], hfeat[(size_t)s * HC + rem] * alpha);
}

// out = relu(in + bias[col]); supports in == out (in-place)
__global__ void bias_relu_kernel(const float* __restrict__ in,
                                 const float* __restrict__ bias,
                                 float* out, long long total, int cols) {
  const long long t = (long long)blockIdx.x * blockDim.x + threadIdx.x;
  if (t >= total) return;
  const float v = in[t] + bias[(int)(t % cols)];
  out[t] = v > 0.f ? v : 0.f;
}

// ---------------------------------------------------------------------------
extern "C" void kernel_launch(void* const* d_in, const int* in_sizes, int n_in,
                              void* d_out, int out_size, void* d_ws, size_t ws_size,
                              hipStream_t stream) {
  const float* x      = (const float*)d_in[0];
  const int*   ei     = (const int*)d_in[1];     // [2,E] flat, int32
  const float* W1     = (const float*)d_in[2];   // [128,256]
  const float* a1_src = (const float*)d_in[3];   // [4,64]
  const float* a1_dst = (const float*)d_in[4];
  const float* b1     = (const float*)d_in[5];   // [256]
  const float* W2     = (const float*)d_in[6];   // [256,64]
  const float* a2_src = (const float*)d_in[7];   // [1,64]
  const float* a2_dst = (const float*)d_in[8];
  const float* b2     = (const float*)d_in[9];   // [64]
  float* out = (float*)d_out;                    // [N,64]

  // ---- workspace carve (256B aligned) ----
  char* ws = (char*)d_ws;
  size_t off = 0;
  auto carve = [&](size_t bytes) -> char* {
    char* p = ws + off;
    off += (bytes + 255) & ~(size_t)255;
    return p;
  };
  const int H1C = GAT_HEADS * GAT_HID;            // 256
  float*    h1    = (float*)   carve((size_t)GAT_N * H1C * 4);
  float*    agg1  = (float*)   carve((size_t)GAT_N * H1C * 4);
  float*    al1s  = (float*)   carve((size_t)GAT_N * GAT_HEADS * 4);
  float*    al1d  = (float*)   carve((size_t)GAT_N * GAT_HEADS * 4);
  int*      emax1 = (int*)     carve((size_t)GAT_N * GAT_HEADS * 4);
  float*    den1  = (float*)   carve((size_t)GAT_N * GAT_HEADS * 4);
  float*    eexp1 = (float*)   carve((size_t)GAT_NE * GAT_HEADS * 4);
  float*    h2    = (float*)   carve((size_t)GAT_N * GAT_HID * 4);
  float*    agg2  = (float*)   carve((size_t)GAT_N * GAT_HID * 4);
  float*    al2s  = (float*)   carve((size_t)GAT_N * 4);
  float*    al2d  = (float*)   carve((size_t)GAT_N * 4);
  int*      emax2 = (int*)     carve((size_t)GAT_N * 4);
  float*    den2  = (float*)   carve((size_t)GAT_N * 4);
  float*    eexp2 = (float*)   carve((size_t)GAT_NE * 4);
  unsigned* xpk   = (unsigned*)carve((size_t)GAT_N * (GAT_INF / 2) * 4);
  unsigned* h2pk  = (unsigned*)carve((size_t)GAT_N * (H1C / 2) * 4);
  unsigned* W1p   = (unsigned*)carve((size_t)(GAT_INF / 2) * H1C * 4);
  unsigned* W2p   = (unsigned*)carve((size_t)(H1C / 2) * GAT_HID * 4);
  (void)ws_size; (void)in_sizes; (void)n_in; (void)out_size;

  // ---- init (graph-capturable) ----
  hipMemsetAsync(den1, 0, (size_t)GAT_N * GAT_HEADS * 4, stream);
  hipMemsetAsync(agg1, 0, (size_t)GAT_N * H1C * 4, stream);
  hipMemsetAsync(den2, 0, (size_t)GAT_N * 4, stream);
  hipMemsetAsync(agg2, 0, (size_t)GAT_N * GAT_HID * 4, stream);
  fill_int<<<(GAT_N * GAT_HEADS + 255) / 256, 256, 0, stream>>>(
      emax1, GAT_N * GAT_HEADS, ENC_NEG_INF);
  fill_int<<<(GAT_N + 255) / 256, 256, 0, stream>>>(emax2, GAT_N, ENC_NEG_INF);

  // ---- bf16 pre-packs (streaming, trivial vs 23.3 TB/s) ----
  {
    const long long np = (long long)GAT_N * (GAT_INF / 2);
    convert_bf16_pairs<<<(int)((np + 255) / 256), 256, 0, stream>>>(x, xpk, np);
  }
  convert_pack_w<<<((GAT_INF / 2) * H1C + 255) / 256, 256, 0, stream>>>(
      W1, W1p, GAT_INF, H1C);
  convert_pack_w<<<((H1C / 2) * GAT_HID + 255) / 256, 256, 0, stream>>>(
      W2, W2p, H1C, GAT_HID);

  // ================= layer 1: 128 -> 4 x 64, concat =================
  gemm_wmma_bf16pk<<<dim3(GAT_N / 80, H1C / 64), 128, 0, stream>>>(
      xpk, W1p, h1, GAT_N, H1C, GAT_INF);
  half_scores<<<GAT_N, GAT_HEADS * 32, 0, stream>>>(
      h1, a1_src, a1_dst, al1s, al1d, H1C, GAT_HID);
  {
    const int nt = GAT_NE * GAT_HEADS;
    edge_max_kernel<<<(nt + 255) / 256, 256, 0, stream>>>(
        ei, al1s, al1d, emax1, GAT_HEADS);
    edge_exp_kernel<<<(nt + 255) / 256, 256, 0, stream>>>(
        ei, al1s, al1d, emax1, eexp1, den1, h1, GAT_HEADS, H1C);
  }
  {
    const long long total = (long long)GAT_NE * H1C;
    aggregate_kernel<<<(int)((total + 255) / 256), 256, 0, stream>>>(
        ei, h1, eexp1, den1, agg1, GAT_HEADS, GAT_HID);
  }
  {
    const long long total = (long long)GAT_N * H1C;
    bias_relu_kernel<<<(int)((total + 255) / 256), 256, 0, stream>>>(
        agg1, b1, agg1, total, H1C);   // in-place: layer-2 input (fp32 view)
  }

  // ================= layer 2: 256 -> 64, 1 head =================
  {
    const long long np = (long long)GAT_N * (H1C / 2);
    convert_bf16_pairs<<<(int)((np + 255) / 256), 256, 0, stream>>>(
        agg1, h2pk, np);
  }
  gemm_wmma_bf16pk<<<dim3(GAT_N / 80, GAT_HID / 64), 128, 0, stream>>>(
      h2pk, W2p, h2, GAT_N, GAT_HID, H1C);
  half_scores<<<GAT_N, 32, 0, stream>>>(
      h2, a2_src, a2_dst, al2s, al2d, GAT_HID, GAT_HID);
  edge_max_kernel<<<(GAT_NE + 255) / 256, 256, 0, stream>>>(
      ei, al2s, al2d, emax2, 1);
  edge_exp_kernel<<<(GAT_NE + 255) / 256, 256, 0, stream>>>(
      ei, al2s, al2d, emax2, eexp2, den2, h2, 1, GAT_HID);
  {
    const long long total = (long long)GAT_NE * GAT_HID;
    aggregate_kernel<<<(int)((total + 255) / 256), 256, 0, stream>>>(
        ei, h2, eexp2, den2, agg2, 1, GAT_HID);
  }
  {
    const long long total = (long long)GAT_N * GAT_HID;
    bias_relu_kernel<<<(int)((total + 255) / 256), 256, 0, stream>>>(
        agg2, b2, out, total, GAT_HID);
  }
}
